// Contrastive_Loss_73169062854885
// MI455X (gfx1250) — matrix operand order repaired and checked by reading, not compiled, not run on hardware
//
#include <hip/hip_runtime.h>

// CDNA5 / gfx1250, wave32. WMMA f16 (K=32) with f32 accumulation,
// LDS-tiled, software-pipelined GEMMs, TDM staging for weight (A) tiles,
// loc+sem encoder convolutions fused into one 512-row GEMM.

typedef __attribute__((ext_vector_type(16))) _Float16 v16h;
typedef __attribute__((ext_vector_type(8)))  _Float16 v8h;
typedef __attribute__((ext_vector_type(8)))  float    v8f;
typedef __attribute__((ext_vector_type(4)))  unsigned int u32x4;
typedef __attribute__((ext_vector_type(8)))  unsigned int u32x8;

#define LDP 40   // padded f16 row stride for 32-wide K tiles (32 + 8)

__device__ __forceinline__ float leaky(float v) { return v > 0.0f ? v : 0.1f * v; }

// ---------------------------------------------------------------------------
// Pre-transpose 3x3 conv weights: (co, ci, kh, kw) -> (co, khw, ci)
// ---------------------------------------------------------------------------
__global__ void transpose_w_kernel(const float* __restrict__ w, float* __restrict__ wt) {
  const int co = blockIdx.x;
  const float* __restrict__ src = w + (size_t)co * 2304;
  float* __restrict__ dst = wt + (size_t)co * 2304;
  for (int i = threadIdx.x; i < 2304; i += blockDim.x) {
    const int khw = i >> 8;     // i = khw*256 + ci
    const int ci  = i & 255;
    dst[i] = src[ci * 9 + khw];
  }
}

__global__ void concat_bias_kernel(const float* __restrict__ a,
                                   const float* __restrict__ b,
                                   float* __restrict__ dst) {
  const int t = threadIdx.x;
  dst[t] = a[t];
  dst[256 + t] = b[t];
}

// ---------------------------------------------------------------------------
// Unified tiled conv GEMM:  out[b, m, n] = sum_k A[m,k] * B[k,n] + bias[m]
//   mode 0 (1x1):  B[k][n] = x[b, k, n], Kdim = Cin, Mtot = 256
//   mode 1 (3x3,s2,p1): Kdim = 2304 (k = khw*256+ci, weights pre-transposed),
//       Mtot = 512 (rows 0..255 = loc encoder, 256..511 = sem encoder)
// Block: 256 threads = 8 waves; block tile 128(M) x 128(N).
// Software pipeline per 32-wide K-step:
//   issue TDM(A,i+1) + B-gather loads(i+1)  ->  8 WMMAs(i)  ->  cvt/store B(i+1)
//   ->  s_wait_tensorcnt + barrier  ->  convert A f32->f16(i+1)  ->  barrier
// ---------------------------------------------------------------------------
__global__ void __launch_bounds__(256) conv_wmma_tiled(
    int mode,
    const float* __restrict__ Aw,   // (Mtot x Kdim) row-major weights
    int Kdim, int Mtot,
    const float* __restrict__ x,
    const float* __restrict__ bias,
    float* __restrict__ out,
    int N, int H, int W, int Wo)
{
  __shared__ float    Af32[128 * 32];        // TDM landing buffer (16 KB)
  __shared__ _Float16 A16[2][128 * LDP];     // f16 A tiles, rows [m][k]
  __shared__ _Float16 B16[2][128 * LDP];     // f16 B tiles, rows [n][k]

  const int t    = threadIdx.x;
  const int lane = t & 31;
  const int wv   = t >> 5;
  const int half = lane >> 4;
  const int l16  = lane & 15;
  const int wm   = wv >> 2;                  // 0..1 : 64-row m slice
  const int wn   = wv & 3;                   // 0..3 : 32-col n slice
  const int mBlk = blockIdx.y * 128;
  const int nBlk = blockIdx.x * 128;
  const int b    = blockIdx.z;

  const size_t xStride = (mode == 0) ? (size_t)Kdim * N : (size_t)256 * H * W;
  const float* __restrict__ xb = x + (size_t)b * xStride;

  // B-gather role: column nS, k-half of 16
  const int nS  = t & 127;
  const int kS  = (t >> 7) * 16;
  const int nG  = nBlk + nS;
  const int nGc = nG < N ? nG : N - 1;
  int oh = 0, ow = 0;
  if (mode == 1) { oh = nGc / Wo; ow = nGc - oh * Wo; }

  // ---- pipeline stage helpers ----
  auto issueTDM = [&](int k0) {              // wave 0 only; EXEC ignored by TDM
    const unsigned long long ga =
        (unsigned long long)(uintptr_t)(Aw + (size_t)mBlk * Kdim + k0);
    const unsigned lds = (unsigned)(uintptr_t)(&Af32[0]);
    u32x4 g0;
    g0[0] = 1u;                                        // count=1, user descriptor
    g0[1] = lds;                                       // lds_addr
    g0[2] = (unsigned)ga;                              // global_addr[31:0]
    g0[3] = (unsigned)((ga >> 32) & 0x01FFFFFFull)     // global_addr[56:32]
            | (2u << 30);                              // type = 2 ("image")
    u32x8 g1;
    g1[0] = 2u << 16;                                  // data_size = 4 bytes
    g1[1] = 0u;                                        // tensor_dim0 lo16 (dim0 = 1<<30)
    g1[2] = 0x4000u;                                   // tensor_dim0 hi16; dim1 lo16
    g1[3] = 0x10u | (32u << 16);                       // dim1 hi16 (1<<20); tile_dim0=32
    g1[4] = 128u;                                      // tile_dim1 = 128
    g1[5] = (unsigned)Kdim;                            // tensor_dim0_stride lo32
    g1[6] = 0u;
    g1[7] = 0u;
    asm volatile("tensor_load_to_lds %0, %1" :: "s"(g0), "s"(g1) : "memory");
  };
  auto gatherB = [&](int k0, float* v) {
    if (mode == 0) {
      const float* __restrict__ p = xb + (size_t)(k0 + kS) * N + nGc;
#pragma unroll
      for (int j = 0; j < 16; ++j) v[j] = p[(size_t)j * N];
    } else {
      const int bk  = k0 + kS;                         // 16-aligned -> khw const
      const int khw = bk >> 8;
      const int kh  = khw / 3;
      const int kw  = khw - kh * 3;
      const int ih  = 2 * oh + kh - 1;
      const int iw  = 2 * ow + kw - 1;
      const bool inb = (ih >= 0) && (ih < H) && (iw >= 0) && (iw < W);
      const int cib = bk & 255;
#pragma unroll
      for (int j = 0; j < 16; ++j) {
        v[j] = 0.0f;
        if (inb) v[j] = xb[((size_t)(cib + j) * H + ih) * W + iw];
      }
    }
  };
  auto cvtStoreB = [&](const float* v, _Float16* Bbuf) {
    _Float16* __restrict__ dst = &Bbuf[nS * LDP + kS];
#pragma unroll
    for (int j = 0; j < 16; ++j) dst[j] = (_Float16)v[j];
  };
  auto convertA = [&](_Float16* Abuf) {      // Af32 -> f16 rows [m][k]
    const int mS = t >> 1;
    const int kT = (t & 1) * 16;
    const float* __restrict__ src = &Af32[mS * 32 + kT];
    _Float16* __restrict__ dst = &Abuf[mS * LDP + kT];
#pragma unroll
    for (int j = 0; j < 16; ++j) dst[j] = (_Float16)src[j];
  };

  v8f acc[4][2];
#pragma unroll
  for (int i = 0; i < 4; ++i)
#pragma unroll
    for (int j = 0; j < 2; ++j) {
      v8f z = {0.f, 0.f, 0.f, 0.f, 0.f, 0.f, 0.f, 0.f};
      acc[i][j] = z;
    }

  const int steps = Kdim / 32;

  // ---- prologue: stage tile 0 ----
  if (t < 32) issueTDM(0);
  {
    float v[16];
    gatherB(0, v);
    cvtStoreB(v, B16[0]);
  }
  if (t < 32) __builtin_amdgcn_s_wait_tensorcnt(0);
  __syncthreads();
  convertA(A16[0]);
  __syncthreads();

  const int akb = half * 8;                  // A lane K-base (+16 split)
  const int bkb = half * 8;                  // B lane K-base (+16 split)

  int p = 0;
  for (int i = 0; i < steps; ++i) {
    const int q = p ^ 1;
    const bool more = (i + 1) < steps;

    // prefetch next tile: TDM for A, global gather for B
    if (more && t < 32) issueTDM((i + 1) * 32);
    float v[16];
    if (more) gatherB((i + 1) * 32, v);

    // ---- WMMA on current tile: wave computes 64(m) x 32(n) ----
    const _Float16* __restrict__ Ab = A16[p];
    const _Float16* __restrict__ Bb = B16[p];
    v16h av[4];
#pragma unroll
    for (int tm = 0; tm < 4; ++tm) {
      const int m = wm * 64 + tm * 16 + l16;
      const v8h lo = *(const v8h*)&Ab[m * LDP + akb];
      const v8h hi = *(const v8h*)&Ab[m * LDP + akb + 16];
#pragma unroll
      for (int e = 0; e < 8; ++e) { av[tm][e] = lo[e]; av[tm][8 + e] = hi[e]; }
    }
#pragma unroll
    for (int tn = 0; tn < 2; ++tn) {
      const int n = wn * 32 + tn * 16 + l16;
      const v8h blo = *(const v8h*)&Bb[n * LDP + bkb];
      const v8h bhi = *(const v8h*)&Bb[n * LDP + bkb + 16];
      v16h bv;
#pragma unroll
      for (int e = 0; e < 8; ++e) { bv[e] = blo[e]; bv[8 + e] = bhi[e]; }
#pragma unroll
      for (int tm = 0; tm < 4; ++tm)
        acc[tm][tn] = __builtin_amdgcn_wmma_f32_16x16x32_f16(
            false, av[tm], false, bv, (short)0, acc[tm][tn], false, false);
    }

    // drain prefetched B into the other buffer (disjoint from compute buffer)
    if (more) cvtStoreB(v, B16[q]);
    if (t < 32) __builtin_amdgcn_s_wait_tensorcnt(0);
    __syncthreads();                          // TDM landed, B16[q] complete
    if (more) convertA(A16[q]);
    __syncthreads();                          // A16[q] ready; Af32 reusable
    p = q;
  }

  // ---- store (+bias), guarded for N tails ----
  float* __restrict__ op = out + (size_t)b * Mtot * N;
#pragma unroll
  for (int tn = 0; tn < 2; ++tn) {
    const int n = nBlk + wn * 32 + tn * 16 + l16;
    if (n < N) {
#pragma unroll
      for (int tm = 0; tm < 4; ++tm) {
        const int mrow = mBlk + wm * 64 + tm * 16 + half * 8;
#pragma unroll
        for (int r = 0; r < 8; ++r)
          op[(size_t)(mrow + r) * N + n] = acc[tm][tn][r] + bias[mrow + r];
      }
    }
  }
}

// ---------------------------------------------------------------------------
// BN batch statistics over (B, H, W) per channel -> fused scale/shift.
// y channel c of batch bb at y + bb*bStride + c*P (supports fused 512-ch buf).
// ---------------------------------------------------------------------------
__global__ void __launch_bounds__(256) bn_stats_kernel(
    const float* __restrict__ y, const float* __restrict__ g,
    const float* __restrict__ beta, int NB, size_t bStride, int P,
    float* __restrict__ scale, float* __restrict__ shift)
{
  const int c = blockIdx.x;
  const int t = threadIdx.x;
  __shared__ float s1[256];
  __shared__ float s2[256];
  float a = 0.0f, b = 0.0f;
  for (int bb = 0; bb < NB; ++bb) {
    const float* __restrict__ p = y + (size_t)bb * bStride + (size_t)c * P;
    for (int i = t; i < P; i += 256) { const float v = p[i]; a += v; b += v * v; }
  }
  s1[t] = a; s2[t] = b;
  __syncthreads();
  for (int s = 128; s > 0; s >>= 1) {
    if (t < s) { s1[t] += s1[t + s]; s2[t] += s2[t + s]; }
    __syncthreads();
  }
  if (t == 0) {
    const float cnt = (float)NB * (float)P;
    const float m   = s1[0] / cnt;
    const float var = s2[0] / cnt - m * m;
    const float r   = rsqrtf(var + 1e-5f);
    const float sc  = g[c] * r;
    scale[c] = sc;
    shift[c] = beta[c] - m * sc;
  }
}

// ---------------------------------------------------------------------------
// fea[b, c] = mean_p leaky(scale[c]*y[b,c,p] + shift[c]); grid = (256, B)
// ---------------------------------------------------------------------------
__global__ void __launch_bounds__(256) bn_pool_kernel(
    const float* __restrict__ y, const float* __restrict__ scale,
    const float* __restrict__ shift, size_t bStride, int P,
    float* __restrict__ feaRow)
{
  const int c = blockIdx.x;
  const int b = blockIdx.y;
  const int t = threadIdx.x;
  __shared__ float s1[256];
  const float* __restrict__ p = y + (size_t)b * bStride + (size_t)c * P;
  const float sc = scale[c], sh = shift[c];
  float s = 0.0f;
  for (int i = t; i < P; i += 256) s += leaky(sc * p[i] + sh);
  s1[t] = s;
  __syncthreads();
  for (int st = 128; st > 0; st >>= 1) {
    if (t < st) s1[t] += s1[t + st];
    __syncthreads();
  }
  if (t == 0) feaRow[b * 256 + c] = s1[0] / (float)P;
}

// ---------------------------------------------------------------------------
// MLP: out = leaky(F @ w1^T + b1) @ w2^T + b2, F is 16x256. WMMA, h in LDS.
// ---------------------------------------------------------------------------
__global__ void __launch_bounds__(256) mlp_wmma(
    const float* __restrict__ F, const float* __restrict__ w1,
    const float* __restrict__ b1, const float* __restrict__ w2,
    const float* __restrict__ b2, float* __restrict__ out)
{
  __shared__ float hbuf[16 * 256];
  const int wv   = threadIdx.x >> 5;
  const int lane = threadIdx.x & 31;
  const int half = lane >> 4;
  const int l16  = lane & 15;

#pragma unroll
  for (int tile = 0; tile < 2; ++tile) {
    const int n0 = (wv * 2 + tile) * 16;
    v8f c = {0.f, 0.f, 0.f, 0.f, 0.f, 0.f, 0.f, 0.f};
    for (int k0 = 0; k0 < 256; k0 += 32) {
      v16h av, bv;
      const int ak = k0 + half * 8;
#pragma unroll
      for (int e = 0; e < 8; ++e) {
        av[e]     = (_Float16)F[l16 * 256 + ak + e];
        av[8 + e] = (_Float16)F[l16 * 256 + ak + 16 + e];
      }
      const int bk = k0 + half * 8;
#pragma unroll
      for (int e = 0; e < 16; ++e)
        bv[e] = (_Float16)w1[(size_t)(n0 + l16) * 256 + bk + e + (e >= 8 ? 8 : 0)];
      c = __builtin_amdgcn_wmma_f32_16x16x32_f16(false, av, false, bv, (short)0, c, false, false);
    }
    const int mrow = half * 8;
#pragma unroll
    for (int r = 0; r < 8; ++r)
      hbuf[(mrow + r) * 256 + n0 + l16] = leaky(c[r] + b1[n0 + l16]);
  }
  __syncthreads();
#pragma unroll
  for (int tile = 0; tile < 2; ++tile) {
    const int n0 = (wv * 2 + tile) * 16;
    v8f c = {0.f, 0.f, 0.f, 0.f, 0.f, 0.f, 0.f, 0.f};
    for (int k0 = 0; k0 < 256; k0 += 32) {
      v16h av, bv;
      const int ak = k0 + half * 8;
#pragma unroll
      for (int e = 0; e < 8; ++e) {
        av[e]     = (_Float16)hbuf[l16 * 256 + ak + e];
        av[8 + e] = (_Float16)hbuf[l16 * 256 + ak + 16 + e];
      }
      const int bk = k0 + half * 8;
#pragma unroll
      for (int e = 0; e < 16; ++e)
        bv[e] = (_Float16)w2[(size_t)(n0 + l16) * 256 + bk + e + (e >= 8 ? 8 : 0)];
      c = __builtin_amdgcn_wmma_f32_16x16x32_f16(false, av, false, bv, (short)0, c, false, false);
    }
    const int mrow = half * 8;
#pragma unroll
    for (int r = 0; r < 8; ++r)
      out[(mrow + r) * 256 + n0 + l16] = c[r] + b2[n0 + l16];
  }
}

// ---------------------------------------------------------------------------
// Contrastive losses. Row layout per encoder (16 x 256):
//   rows 0..7  = Lb/Sb (tb tensors, slot*2+b), rows 8..15 = L/S (xs tensors).
// ---------------------------------------------------------------------------
__global__ void __launch_bounds__(256) loss_kernel(
    const float* __restrict__ outLoc, const float* __restrict__ outSem,
    float* __restrict__ d_out)
{
  __shared__ float R[32 * 256];
  __shared__ float norms[32];
  __shared__ float dots[256];   // [mat(4)][rowA(8)][rowB(8)] of exp(sim/T)
  const int t = threadIdx.x;

  for (int i = t; i < 16 * 256; i += 256) R[i] = outLoc[i];
  for (int i = t; i < 16 * 256; i += 256) R[16 * 256 + i] = outSem[i];
  __syncthreads();

  if (t < 32) {
    float s = 0.0f;
    for (int k = 0; k < 256; ++k) { const float v = R[t * 256 + k]; s += v * v; }
    norms[t] = fmaxf(sqrtf(s), 1e-12f);
  }
  __syncthreads();

  // mat 0: LL (L,L)  1: LLb (L,Lb)  2: SS (S,S)  3: SSb (S,Sb)
  const int mat = t >> 6, idx = t & 63, ra = idx >> 3, rb = idx & 7;
  int rowA, rowB;
  if      (mat == 0) { rowA = 8 + ra;  rowB = 8 + rb;  }
  else if (mat == 1) { rowA = 8 + ra;  rowB = rb;      }
  else if (mat == 2) { rowA = 24 + ra; rowB = 24 + rb; }
  else               { rowA = 24 + ra; rowB = 16 + rb; }
  float s = 0.0f;
  for (int k = 0; k < 256; ++k) s += R[rowA * 256 + k] * R[rowB * 256 + k];
  s = s / (norms[rowA] * norms[rowB]);
  dots[t] = expf(s * 2.0f);     // /TEMP, TEMP = 0.5
  __syncthreads();

  if (t == 0) {                 // geo loss
    float geo = 0.0f;
    for (int ib = 0; ib < 8; ++ib) {
      const int i = ib >> 1, bb = ib & 1;
      const float num = dots[64 + ib * 8 + ib];
      float den = num;
      for (int kl = 0; kl < 8; ++kl) {
        const int k = kl >> 1, l = kl & 1;
        if (i != k && bb != l) den += dots[ib * 8 + kl] + dots[64 + ib * 8 + kl];
      }
      geo -= logf(num / den);
    }
    d_out[0] = geo;
  }
  if (t == 1) {                 // sem loss (rows i < S-1, positive = (i+1, same b))
    float sem = 0.0f;
    for (int i = 0; i < 3; ++i) {
      for (int bb = 0; bb < 2; ++bb) {
        const int ib = i * 2 + bb;
        const float num = dots[128 + ib * 8 + ((i + 1) * 2 + bb)];
        float den = num;
        for (int kl = 0; kl < 8; ++kl) {
          const int l = kl & 1;
          if (l != bb) den += dots[128 + ib * 8 + kl] + dots[192 + ib * 8 + kl];
        }
        sem -= logf(num / den);
      }
    }
    d_out[1] = sem;
  }
}

// ---------------------------------------------------------------------------
extern "C" void kernel_launch(void* const* d_in, const int* in_sizes, int n_in,
                              void* d_out, int out_size, void* d_ws, size_t ws_size,
                              hipStream_t stream) {
  (void)in_sizes; (void)n_in; (void)out_size; (void)ws_size;

  const float* xbi[4] = { (const float*)d_in[0], (const float*)d_in[1],
                          (const float*)d_in[2], (const float*)d_in[3] };
  const float* xs[4]  = { (const float*)d_in[4], (const float*)d_in[5],
                          (const float*)d_in[6], (const float*)d_in[7] };

  // Workspace layout (floats); total ~155 MB.
  float* ws = (float*)d_ws;
  size_t off = 0;
  float* tb0   = ws + off; off += (size_t)2 * 256 * 40000;
  float* tb1   = ws + off; off += (size_t)2 * 256 * 10000;
  float* tb2   = ws + off; off += (size_t)2 * 256 * 2500;
  float* tb3   = ws + off; off += (size_t)2 * 256 * 625;
  float* wtAll = ws + off; off += (size_t)512 * 2304;        // [loc;sem] transposed
  float* cbAll = ws + off; off += 512;
  float* ybuf  = ws + off; off += (size_t)2 * 512 * 10000;   // fused conv3x3 out
  float* scale = ws + off; off += 256;
  float* shift = ws + off; off += 256;
  float* feaLoc = ws + off; off += 16 * 256;
  float* feaSem = ws + off; off += 16 * 256;
  float* outLoc = ws + off; off += 16 * 256;
  float* outSem = ws + off; off += 16 * 256;

  // 1. transpose loc/sem 3x3 weights into one 512 x 2304 matrix + fused bias
  transpose_w_kernel<<<256, 256, 0, stream>>>((const float*)d_in[16], wtAll);
  transpose_w_kernel<<<256, 256, 0, stream>>>((const float*)d_in[24],
                                              wtAll + (size_t)256 * 2304);
  concat_bias_kernel<<<1, 256, 0, stream>>>((const float*)d_in[17],
                                            (const float*)d_in[25], cbAll);

  // 2. conv1x1 projections (tiled WMMA GEMM, TDM-staged weights)
  const int HW[4]  = { 40000, 10000, 2500, 625 };
  const int Cin[4] = { 256, 512, 1024, 2048 };
  float* tb[4] = { tb0, tb1, tb2, tb3 };
  for (int i = 0; i < 4; ++i) {
    dim3 g((HW[i] + 127) / 128, 2, 2);
    conv_wmma_tiled<<<g, 256, 0, stream>>>(0, (const float*)d_in[8 + 2 * i],
                                           Cin[i], 256, xbi[i],
                                           (const float*)d_in[9 + 2 * i],
                                           tb[i], HW[i], 0, 0, 0);
  }

  // 3. 8 fused encoder passes: conv3x3 (loc+sem) -> per-encoder bn -> pool
  const int Hs[8] = { 200, 100, 50, 25, 200, 100, 50, 25 };
  const float* src[8] = { tb0, tb1, tb2, tb3, xs[0], xs[1], xs[2], xs[3] };
  for (int ti = 0; ti < 8; ++ti) {
    const int H = Hs[ti], W = H;
    const int Ho = (H - 1) / 2 + 1, Wo = Ho;
    const int N = Ho * Wo;
    dim3 gc((N + 127) / 128, 4, 2);           // Mtot = 512 rows
    conv_wmma_tiled<<<gc, 256, 0, stream>>>(1, wtAll, 2304, 512, src[ti],
                                            cbAll, ybuf, N, H, W, Wo);
    const size_t bStride = (size_t)512 * N;
    for (int e = 0; e < 2; ++e) {
      const float* gg = (const float*)d_in[e ? 26 : 18];
      const float* be = (const float*)d_in[e ? 27 : 19];
      float* fea = e ? feaSem : feaLoc;
      const float* ye = ybuf + (size_t)e * 256 * N;
      bn_stats_kernel<<<256, 256, 0, stream>>>(ye, gg, be, 2, bStride, N,
                                               scale, shift);
      dim3 gp(256, 2);
      bn_pool_kernel<<<gp, 256, 0, stream>>>(ye, scale, shift, bStride, N,
                                             fea + ti * 512);
    }
  }

  // 4. MLP heads (WMMA)
  mlp_wmma<<<1, 256, 0, stream>>>(feaLoc, (const float*)d_in[20], (const float*)d_in[21],
                                  (const float*)d_in[22], (const float*)d_in[23], outLoc);
  mlp_wmma<<<1, 256, 0, stream>>>(feaSem, (const float*)d_in[28], (const float*)d_in[29],
                                  (const float*)d_in[30], (const float*)d_in[31], outSem);

  // 5. losses
  loss_kernel<<<1, 256, 0, stream>>>(outLoc, outSem, (float*)d_out);
}
